// TransformerBlock_29171417875245
// MI455X (gfx1250) — matrix operand order repaired and checked by reading, not compiled, not run on hardware
//
#include <hip/hip_runtime.h>

#define D_MODEL 1024
#define NHEAD   16
#define HEADD   64
#define FF      4096
#define NE      8
#define BATCH   4
#define SEQ     2048
#define NTOK    (BATCH*SEQ)
#define KSP     614   // int(2048*0.3)

typedef __attribute__((ext_vector_type(16))) __bf16 v16bf;
typedef __attribute__((ext_vector_type(8)))  float  v8f;
typedef __attribute__((ext_vector_type(4)))  unsigned int u32x4;
typedef __attribute__((ext_vector_type(8)))  int  i32x8;
typedef __attribute__((ext_vector_type(4)))  int  i32x4;

union Frag16 { v16bf v; uint4 u[2]; };

// low 32 bits of a generic pointer into LDS == wave-relative LDS byte address
#define LDS_OFF(p) ((unsigned)(unsigned long long)(p))

__device__ __forceinline__ unsigned short f2bf(float f) {
  union { float f; unsigned u; } x; x.f = f;
  unsigned r = x.u + 0x7fffu + ((x.u >> 16) & 1u);
  return (unsigned short)(r >> 16);
}
__device__ __forceinline__ float bf2f(unsigned short h) {
  union { unsigned u; float f; } x; x.u = ((unsigned)h) << 16; return x.f;
}

// async global->LDS 128b copy (ASYNCcnt tracked); EXEC-masked per lane
__device__ __forceinline__ void async_b128(unsigned ldsAddr, const void* gaddr) {
  asm volatile("global_load_async_to_lds_b128 %0, %1, off"
               :: "v"(ldsAddr), "v"((unsigned long long)gaddr) : "memory");
}
__device__ __forceinline__ void wait_async0() {
  asm volatile("s_wait_asynccnt 0x0" ::: "memory");
}

// ---------------------------------------------------------------- convert
__global__ void f32_to_bf16_kernel(const float* __restrict__ src,
                                   unsigned short* __restrict__ dst, int n) {
  int i = blockIdx.x * 256 + threadIdx.x;
  if (i < n) dst[i] = f2bf(src[i]);
}

// ---------------------------------------------------------------- layernorm
__global__ void ln_bf16_kernel(const float* __restrict__ x,
                               const float* __restrict__ g,
                               const float* __restrict__ b,
                               unsigned short* __restrict__ out) {
  const int t = blockIdx.x;
  const int tid = threadIdx.x;
  __shared__ float red[256];
  const float* xr = x + (size_t)t * D_MODEL;
  float s = 0.f;
  for (int i = tid; i < D_MODEL; i += 256) s += xr[i];
  red[tid] = s; __syncthreads();
  for (int o = 128; o > 0; o >>= 1) { if (tid < o) red[tid] += red[tid + o]; __syncthreads(); }
  float mu = red[0] * (1.f / D_MODEL);
  __syncthreads();
  float vv = 0.f;
  for (int i = tid; i < D_MODEL; i += 256) { float d = xr[i] - mu; vv += d * d; }
  red[tid] = vv; __syncthreads();
  for (int o = 128; o > 0; o >>= 1) { if (tid < o) red[tid] += red[tid + o]; __syncthreads(); }
  float rstd = rsqrtf(red[0] * (1.f / D_MODEL) + 1e-5f);
  for (int i = tid; i < D_MODEL; i += 256)
    out[(size_t)t * D_MODEL + i] = f2bf((xr[i] - mu) * rstd * g[i] + b[i]);
}

// ---------------------------------------------------------------- generic WMMA GEMM
// C[M,N] = act(A[M,K](bf16) @ B[K,N](bf16) + bias) (+resid)
// EXP==0 normal, EXP==1 gathered rows (tok list), EXP==2 direct rows at expBase[e]
template<int ACT, bool OUTBF, int EXP, bool RESID>
__global__ void __launch_bounds__(256) gemm_bf16(
    const unsigned short* __restrict__ A, int lda,
    const unsigned short* __restrict__ Bw, int ldb,
    const float* __restrict__ bias,
    const float* __restrict__ resid,
    void* __restrict__ Cout, int ldc,
    int M, int N, int K,
    const int* __restrict__ rowIdx,
    const int* __restrict__ expBase,
    const int* __restrict__ expCnt)
{
  const int lane = threadIdx.x & 31;
  const int wave = threadIdx.x >> 5;
  const int lo = lane & 15, hi = lane >> 4;
  const int wm = wave >> 1, wn = wave & 1;

  int rowStart = 0, Mloc = M;
  const unsigned short* Bp = Bw;
  const float* biasP = bias;
  if (EXP) {
    int e = blockIdx.z;
    Bp += (size_t)e * K * N;
    biasP += (size_t)e * N;
    rowStart = expBase[e];
    Mloc = expCnt[e];
  }
  const int row0 = blockIdx.y * 64;
  if (row0 >= Mloc) return;
  const int col0 = blockIdx.x * 128;

  __shared__ __align__(16) unsigned short As[64 * 32];      // [m][k]
  __shared__ __align__(16) unsigned short Bst[128 * 32];    // [n][k] transposed

  // A-row for this thread's staging slot (rows beyond Mloc never issue an
  // async copy; their LDS rows only feed C-rows the epilogue masks out)
  const int sm  = threadIdx.x >> 2, sseg = threadIdx.x & 3;
  const int sr  = row0 + sm;
  int arow = 0;
  if (sr < Mloc) {
    if (EXP == 1)      arow = rowIdx[rowStart + sr];
    else if (EXP == 2) arow = rowStart + sr;
    else               arow = sr;
  }
  const unsigned ldsA = LDS_OFF(As + sm * 32 + sseg * 8);

  v8f acc[4];
  #pragma unroll
  for (int t = 0; t < 4; ++t)
    #pragma unroll
    for (int r = 0; r < 8; ++r) acc[t][r] = 0.f;

  for (int k0 = 0; k0 < K; k0 += 32) {
    // stage A tile 64x32: async global->LDS, 1 x B128 per lane
    if (sr < Mloc)
      async_b128(ldsA, A + (size_t)arow * lda + k0 + sseg * 8);
    #pragma unroll
    for (int it = 0; it < 2; ++it) { // stage B tile 32x128 -> transposed LDS
      int lin = threadIdx.x + it * 256;
      int kk = lin >> 4;
      int nn = (lin & 15) * 8;
      int gn = col0 + nn;
      uint4 v = {0u, 0u, 0u, 0u};
      if (gn < N) {
        const unsigned short* bp = Bp + (size_t)(k0 + kk) * ldb + gn;
        v = *(const uint4*)bp;
        if (k0 + 32 < K) __builtin_prefetch(bp + (size_t)32 * ldb, 0, 1);
      }
      const unsigned short* pv = (const unsigned short*)&v;
      #pragma unroll
      for (int j = 0; j < 8; ++j) Bst[(nn + j) * 32 + kk] = pv[j];
    }
    wait_async0();
    __syncthreads();

    // A fragment: lane lo = row (wm*16+lo); contiguous K run of 16 per lane half
    Frag16 a;
    a.u[0] = *(const uint4*)(As + (wm * 16 + lo) * 32 + hi * 16);
    a.u[1] = *(const uint4*)(As + (wm * 16 + lo) * 32 + hi * 16 + 8);
    #pragma unroll
    for (int t = 0; t < 4; ++t) {
      Frag16 bf;
      const unsigned short* bp = Bst + (wn * 64 + t * 16 + lo) * 32 + hi * 16;
      bf.u[0] = ((const uint4*)bp)[0];
      bf.u[1] = ((const uint4*)bp)[1];
      acc[t] = __builtin_amdgcn_wmma_f32_16x16x32_bf16(
          false, a.v, false, bf.v, (short)0, acc[t], false, false);
    }
    __syncthreads();
  }

  #pragma unroll
  for (int t = 0; t < 4; ++t) {
    int gc = col0 + wn * 64 + t * 16 + lo;
    if (gc >= N) continue;
    float bv = biasP ? biasP[gc] : 0.f;
    #pragma unroll
    for (int r = 0; r < 8; ++r) {
      int gm = row0 + wm * 16 + r + 8 * hi;
      if (gm >= Mloc) continue;
      float v = acc[t][r] + bv;
      if (ACT == 1) v = fmaxf(v, 0.f);
      if (ACT == 2) v = 0.5f * v * (1.f + erff(v * 0.70710678118f));
      if (RESID) v += resid[(size_t)gm * ldc + gc];
      size_t orow = (size_t)(EXP ? (rowStart + gm) : gm);
      if (OUTBF) ((unsigned short*)Cout)[orow * ldc + gc] = f2bf(v);
      else       ((float*)Cout)[orow * ldc + gc] = v;
    }
  }
}

// ---------------------------------------------------------------- importance dot (512 -> 1)
__global__ void importance_kernel(const unsigned short* __restrict__ ih,
                                  const float* __restrict__ w2,
                                  const float* __restrict__ b2,
                                  float* __restrict__ imp) {
  const int t = blockIdx.x;
  __shared__ float red[128];
  float s = 0.f;
  for (int i = threadIdx.x; i < 512; i += 128) s += bf2f(ih[(size_t)t * 512 + i]) * w2[i];
  red[threadIdx.x] = s; __syncthreads();
  for (int o = 64; o > 0; o >>= 1) { if (threadIdx.x < o) red[threadIdx.x] += red[threadIdx.x + o]; __syncthreads(); }
  if (threadIdx.x == 0) imp[t] = red[0] + b2[0];
}

// ---------------------------------------------------------------- top-k sparse mask (rank count)
__global__ void topk_mask_kernel(const float* __restrict__ imp,
                                 float* __restrict__ maskadd) {
  const int b = blockIdx.x;
  __shared__ float vals[SEQ];
  for (int i = threadIdx.x; i < SEQ; i += 256) vals[i] = imp[b * SEQ + i];
  __syncthreads();
  for (int i = threadIdx.x; i < SEQ; i += 256) {
    float v = vals[i];
    int rank = 0;
    for (int j = 0; j < SEQ; ++j) {
      float u = vals[j];
      rank += (u > v) || (u == v && j < i);   // stable tie-break like lax.top_k
    }
    maskadd[b * SEQ + i] = (rank < KSP) ? 0.f : -1e30f;
  }
}

// ---------------------------------------------------------------- flash attention (sparse keys)
__global__ void __launch_bounds__(128) attn_kernel(
    const unsigned short* __restrict__ Qb,
    const unsigned short* __restrict__ Kb,
    const unsigned short* __restrict__ Vb,
    const float* __restrict__ maskadd,
    unsigned short* __restrict__ Ctx)
{
  const int bh = blockIdx.x;
  const int b = bh >> 4, h = bh & 15;
  const int wave = threadIdx.x >> 5;       // 4 waves
  const int lane = threadIdx.x & 31;
  const int lo = lane & 15, hi = lane >> 4;
  const int q0 = blockIdx.y * 64 + wave * 16;
  const int btok = b * SEQ;

  __shared__ __align__(16) unsigned short Ks[32 * 64];   // [kv][hd] (TDM dest)
  __shared__ __align__(16) unsigned short Vt[64 * 32];   // [hd][kv]
  __shared__ __align__(16) unsigned short Ps[4][16 * 32];

  Frag16 qf[2];
  {
    const unsigned short* qrow = Qb + (size_t)(btok + q0 + lo) * D_MODEL + h * HEADD;
    #pragma unroll
    for (int c = 0; c < 2; ++c) {
      qf[c].u[0] = *(const uint4*)(qrow + c * 32 + hi * 16);
      qf[c].u[1] = *(const uint4*)(qrow + c * 32 + hi * 16 + 8);
    }
  }

  // --- Tensor DMA descriptor (D#) for the 32(kv) x 64(hd) bf16 K tile -----
  // group1: data_size=2B, tile_dim0=64, tile_dim1=32, tensor_dim0_stride=1024
  const unsigned ldsK = LDS_OFF(Ks);
  const unsigned TD0 = 1u << 30, TD1 = 1u << 30;      // huge tensor dims (tiles in-bounds)
  const i32x8 dg1 = {
      (int)(1u << 16),                                  // data_size code 1 (2 bytes)
      (int)((TD0 & 0xFFFFu) << 16),                     // tensor_dim0[15:0]
      (int)(((TD0 >> 16) & 0xFFFFu) | ((TD1 & 0xFFFFu) << 16)),
      (int)(((TD1 >> 16) & 0xFFFFu) | (64u << 16)),     // tile_dim0 = 64
      (int)(32u),                                       // tile_dim1 = 32, tile_dim2 = 0
      (int)(D_MODEL),                                   // tensor_dim0_stride = 1024
      0, 0 };
  const i32x4 dzero4 = {0, 0, 0, 0};
  const i32x8 dzero8 = {0, 0, 0, 0, 0, 0, 0, 0};

  float m8[8], l8[8];
  v8f acc[4];
  #pragma unroll
  for (int r = 0; r < 8; ++r) { m8[r] = -3.0e38f; l8[r] = 0.f; }
  #pragma unroll
  for (int t = 0; t < 4; ++t)
    #pragma unroll
    for (int r = 0; r < 8; ++r) acc[t][r] = 0.f;

  for (int kv0 = 0; kv0 < SEQ; kv0 += 32) {
    // K tile via Tensor Data Mover (one DMA per block, wave 0 issues)
    if (wave == 0) {
      unsigned long long ga =
          (unsigned long long)(Kb + (size_t)(btok + kv0) * D_MODEL + h * HEADD);
      const u32x4 dg0 = {
          1u,                                           // count=1, user descriptor
          ldsK,                                         // lds_addr
          (unsigned)(ga & 0xFFFFFFFFu),                 // global_addr[31:0]
          (unsigned)(((ga >> 32) & 0x01FFFFFFu) | (2u << 30)) }; // addr[56:32] | type=2
      __builtin_amdgcn_tensor_load_to_lds(dg0, dg1, dzero4, dzero4, dzero8, 0);
    }
    // V tile staged transposed (TDM cannot transpose into [hd][kv])
    #pragma unroll
    for (int it = 0; it < 2; ++it) {
      int lin = threadIdx.x + it * 128;
      int row = lin >> 3, seg = lin & 7;
      uint4 vv4 = *(const uint4*)(Vb + (size_t)(btok + kv0 + row) * D_MODEL + h * HEADD + seg * 8);
      const unsigned short* pv = (const unsigned short*)&vv4;
      #pragma unroll
      for (int j = 0; j < 8; ++j) Vt[(seg * 8 + j) * 32 + row] = pv[j];
    }
    if (wave == 0) __builtin_amdgcn_s_wait_tensorcnt((short)0);
    __syncthreads();

    v8f s[2];
    #pragma unroll
    for (int t = 0; t < 2; ++t) {
      #pragma unroll
      for (int r = 0; r < 8; ++r) s[t][r] = 0.f;
      #pragma unroll
      for (int c = 0; c < 2; ++c) {
        Frag16 kf;
        const unsigned short* kp = Ks + (t * 16 + lo) * 64 + c * 32 + hi * 16;
        kf.u[0] = ((const uint4*)kp)[0];
        kf.u[1] = ((const uint4*)kp)[1];
        s[t] = __builtin_amdgcn_wmma_f32_16x16x32_bf16(
            false, qf[c].v, false, kf.v, (short)0, s[t], false, false);
      }
    }
    const float mk0 = maskadd[btok + kv0 + lo];
    const float mk1 = maskadd[btok + kv0 + 16 + lo];

    #pragma unroll
    for (int r = 0; r < 8; ++r) {
      float a0 = s[0][r] * 0.125f + mk0;
      float a1 = s[1][r] * 0.125f + mk1;
      float mx = fmaxf(a0, a1);
      mx = fmaxf(mx, __shfl_xor(mx, 1, 16));
      mx = fmaxf(mx, __shfl_xor(mx, 2, 16));
      mx = fmaxf(mx, __shfl_xor(mx, 4, 16));
      mx = fmaxf(mx, __shfl_xor(mx, 8, 16));
      float mnew = fmaxf(m8[r], mx);
      float alpha = __expf(m8[r] - mnew);
      float p0 = __expf(a0 - mnew);
      float p1 = __expf(a1 - mnew);
      float rs = p0 + p1;
      rs += __shfl_xor(rs, 1, 16);
      rs += __shfl_xor(rs, 2, 16);
      rs += __shfl_xor(rs, 4, 16);
      rs += __shfl_xor(rs, 8, 16);
      l8[r] = l8[r] * alpha + rs;
      m8[r] = mnew;
      acc[0][r] *= alpha; acc[1][r] *= alpha; acc[2][r] *= alpha; acc[3][r] *= alpha;
      unsigned short* pw = &Ps[wave][(r + 8 * hi) * 32];
      pw[lo]      = f2bf(p0);
      pw[16 + lo] = f2bf(p1);
    }
    __syncthreads();
    Frag16 pf;
    pf.u[0] = *(const uint4*)(&Ps[wave][lo * 32 + hi * 16]);
    pf.u[1] = *(const uint4*)(&Ps[wave][lo * 32 + hi * 16 + 8])
    #pragma unroll
    for (int t2 = 0; t2 < 4; ++t2) {
      Frag16 vf;
      const unsigned short* vp = Vt + (t2 * 16 + lo) * 32 + hi * 16;
      vf.u[0] = ((const uint4*)vp)[0];
      vf.u[1] = ((const uint4*)vp)[1];
      acc[t2] = __builtin_amdgcn_wmma_f32_16x16x32_bf16(
          false, pf.v, false, vf.v, (short)0, acc[t2], false, false);
    }
    __syncthreads();
  }

  #pragma unroll
  for (int r = 0; r < 8; ++r) {
    float inv = 1.f / l8[r];
    int tok = btok + q0 + r + 8 * hi;
    #pragma unroll
    for (int t2 = 0; t2 < 4; ++t2)
      Ctx[(size_t)tok * D_MODEL + h * HEADD + t2 * 16 + lo] = f2bf(acc[t2][r] * inv);
  }
}

// ---------------------------------------------------------------- MoE routing
__global__ void zero_cnt_kernel(int* p) { if (threadIdx.x < 8) p[threadIdx.x] = 0; }

__global__ void route_kernel(const unsigned short* __restrict__ xn2,
                             const float* __restrict__ gw,
                             const float* __restrict__ gb,
                             int* __restrict__ re, float* __restrict__ rw,
                             int* __restrict__ cnt) {
  const int t = blockIdx.x;
  __shared__ float accs[NE];
  if (threadIdx.x < NE) accs[threadIdx.x] = gb[threadIdx.x];
  __syncthreads();
  float p[NE];
  #pragma unroll
  for (int e = 0; e < NE; ++e) p[e] = 0.f;
  for (int d = threadIdx.x; d < D_MODEL; d += 256) {
    float xv = bf2f(xn2[(size_t)t * D_MODEL + d]);
    const float* gr = gw + d * NE;
    #pragma unroll
    for (int e = 0; e < NE; ++e) p[e] += xv * gr[e];
  }
  #pragma unroll
  for (int e = 0; e < NE; ++e) atomicAdd(&accs[e], p[e]);
  __syncthreads();
  if (threadIdx.x == 0) {
    float mx = accs[0];
    for (int e = 1; e < NE; ++e) mx = fmaxf(mx, accs[e]);
    float pr[NE];
    for (int e = 0; e < NE; ++e) pr[e] = __expf(accs[e] - mx);
    int e0 = 0;
    for (int e = 1; e < NE; ++e) if (pr[e] > pr[e0]) e0 = e;
    int e1 = (e0 == 0) ? 1 : 0;
    for (int e = 0; e < NE; ++e) if (e != e0 && pr[e] > pr[e1]) e1 = e;
    float inv = 1.f / (pr[e0] + pr[e1]);
    re[t * 2] = e0; re[t * 2 + 1] = e1;
    rw[t * 2] = pr[e0] * inv; rw[t * 2 + 1] = pr[e1] * inv;
    atomicAdd(&cnt[e0], 1); atomicAdd(&cnt[e1], 1);
  }
}

__global__ void scan_kernel(const int* cnt, int* base, int* cur) {
  if (threadIdx.x == 0) {
    int s = 0;
    for (int e = 0; e < NE; ++e) { base[e] = s; s += cnt[e]; cur[e] = 0; }
  }
}

__global__ void assign_kernel(const int* __restrict__ re,
                              const int* __restrict__ base, int* __restrict__ cur,
                              int* __restrict__ tok, int* __restrict__ slot) {
  int t = blockIdx.x * 256 + threadIdx.x;
  if (t >= NTOK) return;
  #pragma unroll
  for (int k = 0; k < 2; ++k) {
    int e = re[t * 2 + k];
    int pos = base[e] + atomicAdd(&cur[e], 1);
    tok[pos] = t;
    slot[t * 2 + k] = pos;
  }
}

__global__ void combine_kernel(const float* __restrict__ x1,
                               const float* __restrict__ y,
                               const int* __restrict__ slot,
                               const float* __restrict__ rw,
                               float* __restrict__ out) {
  const int t = blockIdx.x;
  const int s0 = slot[t * 2], s1 = slot[t * 2 + 1];
  const float w0 = rw[t * 2], w1 = rw[t * 2 + 1];
  for (int i = threadIdx.x; i < D_MODEL; i += 256)
    out[(size_t)t * D_MODEL + i] = x1[(size_t)t * D_MODEL + i]
        + w0 * y[(size_t)s0 * D_MODEL + i]
        + w1 * y[(size_t)s1 * D_MODEL + i];
}

// ---------------------------------------------------------------- launch
extern "C" void kernel_launch(void* const* d_in, const int* in_sizes, int n_in,
                              void* d_out, int out_size, void* d_ws, size_t ws_size,
                              hipStream_t stream) {
  const float* x     = (const float*)d_in[0];
  const float* ln1_g = (const float*)d_in[1];
  const float* ln1_b = (const float*)d_in[2];
  const float* ln2_g = (const float*)d_in[3];
  const float* ln2_b = (const float*)d_in[4];
  const float* q_w = (const float*)d_in[5];   const float* q_b = (const float*)d_in[6];
  const float* k_w = (const float*)d_in[7];   const float* k_b = (const float*)d_in[8];
  const float* v_w = (const float*)d_in[9];   const float* v_b = (const float*)d_in[10];
  const float* o_w = (const float*)d_in[11];  const float* o_b = (const float*)d_in[12];
  const float* ix_w1 = (const float*)d_in[13]; const float* ix_b1 = (const float*)d_in[14];
  const float* ix_w2 = (const float*)d_in[15]; const float* ix_b2 = (const float*)d_in[16];
  const float* gate_w = (const float*)d_in[17]; const float* gate_b = (const float*)d_in[18];
  const float* e_w1 = (const float*)d_in[19]; const float* e_b1 = (const float*)d_in[20];
  const float* e_w2 = (const float*)d_in[21]; const float* e_b2 = (const float*)d_in[22];

  char* w = (char*)d_ws;
  size_t off = 0;
  auto alloc = [&](size_t bytes) -> void* {
    void* p = w + off;
    off += (bytes + 255) & ~(size_t)255;
    return p;
  };

  unsigned short* xn   = (unsigned short*)alloc((size_t)NTOK * D_MODEL * 2);
  unsigned short* qb   = (unsigned short*)alloc((size_t)NTOK * D_MODEL * 2);
  unsigned short* kb   = (unsigned short*)alloc((size_t)NTOK * D_MODEL * 2);
  unsigned short* vb   = (unsigned short*)alloc((size_t)NTOK * D_MODEL * 2);
  unsigned short* ctx  = (unsigned short*)alloc((size_t)NTOK * D_MODEL * 2);
  unsigned short* xn2  = (unsigned short*)alloc((size_t)NTOK * D_MODEL * 2);
  unsigned short* ih   = (unsigned short*)alloc((size_t)NTOK * 512 * 2);
  float* imp     = (float*)alloc((size_t)NTOK * 4);
  float* maskadd = (float*)alloc((size_t)NTOK * 4);
  float* x1      = (float*)alloc((size_t)NTOK * D_MODEL * 4);
  int*   re      = (int*)alloc((size_t)NTOK * 2 * 4);
  float* rw      = (float*)alloc((size_t)NTOK * 2 * 4);
  int*   meta    = (int*)alloc(256);              // cnt[8], base[8], cur[8]
  int* cnt = meta, *base = meta + 8, *cur = meta + 16;
  int*   tok     = (int*)alloc((size_t)2 * NTOK * 4);
  int*   slot    = (int*)alloc((size_t)2 * NTOK * 4);
  unsigned short* hbuf = (unsigned short*)alloc((size_t)2 * NTOK * FF * 2);
  float* ybuf    = (float*)alloc((size_t)2 * NTOK * D_MODEL * 4);
  unsigned short* qwb = (unsigned short*)alloc((size_t)D_MODEL * D_MODEL * 2);
  unsigned short* kwb = (unsigned short*)alloc((size_t)D_MODEL * D_MODEL * 2);
  unsigned short* vwb = (unsigned short*)alloc((size_t)D_MODEL * D_MODEL * 2);
  unsigned short* owb = (unsigned short*)alloc((size_t)D_MODEL * D_MODEL * 2);
  unsigned short* ixw1b = (unsigned short*)alloc((size_t)D_MODEL * 512 * 2);
  unsigned short* ew1b = (unsigned short*)alloc((size_t)NE * D_MODEL * FF * 2);
  unsigned short* ew2b = (unsigned short*)alloc((size_t)NE * FF * D_MODEL * 2);
  (void)ws_size; (void)n_in; (void)in_sizes; (void)out_size;

  auto conv = [&](const float* s, unsigned short* d, int n) {
    f32_to_bf16_kernel<<<(n + 255) / 256, 256, 0, stream>>>(s, d, n);
  };
  conv(q_w, qwb, D_MODEL * D_MODEL);
  conv(k_w, kwb, D_MODEL * D_MODEL);
  conv(v_w, vwb, D_MODEL * D_MODEL);
  conv(o_w, owb, D_MODEL * D_MODEL);
  conv(ix_w1, ixw1b, D_MODEL * 512);
  conv(e_w1, ew1b, NE * D_MODEL * FF);
  conv(e_w2, ew2b, NE * FF * D_MODEL);

  // LN1
  ln_bf16_kernel<<<NTOK, 256, 0, stream>>>(x, ln1_g, ln1_b, xn);

  // Q/K/V projections (bf16 out)
  dim3 gDD(D_MODEL / 128, NTOK / 64, 1);
  gemm_bf16<0, true, 0, false><<<gDD, 256, 0, stream>>>(
      xn, D_MODEL, qwb, D_MODEL, q_b, nullptr, qb, D_MODEL, NTOK, D_MODEL, D_MODEL, nullptr, nullptr, nullptr);
  gemm_bf16<0, true, 0, false><<<gDD, 256, 0, stream>>>(
      xn, D_MODEL, kwb, D_MODEL, k_b, nullptr, kb, D_MODEL, NTOK, D_MODEL, D_MODEL, nullptr, nullptr, nullptr);
  gemm_bf16<0, true, 0, false><<<gDD, 256, 0, stream>>>(
      xn, D_MODEL, vwb, D_MODEL, v_b, nullptr, vb, D_MODEL, NTOK, D_MODEL, D_MODEL, nullptr, nullptr, nullptr);

  // importance MLP: relu(xn @ ix_w1 + b1) then dot with ix_w2
  dim3 gIH(512 / 128, NTOK / 64, 1);
  gemm_bf16<1, true, 0, false><<<gIH, 256, 0, stream>>>(
      xn, D_MODEL, ixw1b, 512, ix_b1, nullptr, ih, 512, NTOK, 512, D_MODEL, nullptr, nullptr, nullptr);
  importance_kernel<<<NTOK, 128, 0, stream>>>(ih, ix_w2, ix_b2, imp);
  topk_mask_kernel<<<BATCH, 256, 0, stream>>>(imp, maskadd);

  // flash attention (K tile via TDM, sparse-key additive mask)
  attn_kernel<<<dim3(BATCH * NHEAD, SEQ / 64, 1), 128, 0, stream>>>(qb, kb, vb, maskadd, ctx);

  // O projection + residual (fp32 out)
  gemm_bf16<0, false, 0, true><<<gDD, 256, 0, stream>>>(
      ctx, D_MODEL, owb, D_MODEL, o_b, x, x1, D_MODEL, NTOK, D_MODEL, D_MODEL, nullptr, nullptr, nullptr);

  // LN2 + routing
  ln_bf16_kernel<<<NTOK, 256, 0, stream>>>(x1, ln2_g, ln2_b, xn2);
  zero_cnt_kernel<<<1, 32, 0, stream>>>(cnt);
  route_kernel<<<NTOK, 256, 0, stream>>>(xn2, gate_w, gate_b, re, rw, cnt);
  scan_kernel<<<1, 32, 0, stream>>>(cnt, base, cur);
  assign_kernel<<<NTOK / 256, 256, 0, stream>>>(re, base, cur, tok, slot);

  // expert FFN (only top-2 assignments computed; grids worst-case, early-exit on counts)
  dim3 gE1(FF / 128, (NTOK / 64), NE);       // up to 8192 rows per expert
  gemm_bf16<2, true, 1, false><<<gE1, 256, 0, stream>>>(
      xn2, D_MODEL, ew1b, FF, e_b1, nullptr, hbuf, FF, 0, FF, D_MODEL, tok, base, cnt);
  dim3 gE2(D_MODEL / 128, (NTOK / 64), NE);
  gemm_bf16<0, false, 2, false><<<gE2, 256, 0, stream>>>(
      hbuf, FF, ew2b, D_MODEL, e_b2, nullptr, ybuf, D_MODEL, 0, D_MODEL, FF, nullptr, base, cnt);

  // combine: out = x1 + w0*y[slot0] + w1*y[slot1]
  combine_kernel<<<NTOK, 256, 0, stream>>>(x1, ybuf, slot, rw, (float*)d_out);
}
